// CosformerSelfAttention_3040836845928
// MI455X (gfx1250) — matrix operand order repaired and checked by reading, not compiled, not run on hardware
//
#include <hip/hip_runtime.h>
#include <hip/hip_bf16.h>

// ---------------- problem constants ----------------
#define BB 4
#define LL 4096
#define EE 1024
#define HH 16
#define DD 64
#define NN (BB * HH)   // 64 heads
#define FF (2 * DD)    // 128 feature dim
#define EPS 1e-6f
#define PI_HALF 1.57079632679489662f

// pass-1 tiling
#define SEQ_WG 512     // seq positions per workgroup
#define CH 64          // seq positions staged in LDS per iteration
#define KSTR 72        // padded seq stride (halves) for pass1 LDS rows

// pass-2 tiling
#define ROWS 128       // q rows per workgroup
#define QSTR 136       // padded feature stride (halves) for pass2 LDS rows

typedef __attribute__((ext_vector_type(16))) __bf16 v16bf;
typedef __attribute__((ext_vector_type(8)))  float  v8f;

union ABFrag { v16bf v; unsigned int u[8]; };

__device__ __forceinline__ unsigned short f2bf(float f) {
  unsigned int u = __float_as_uint(f);
  unsigned int r = u + 0x7FFFu + ((u >> 16) & 1u);  // round-to-nearest-even
  return (unsigned short)(r >> 16);
}
__device__ __forceinline__ float bf2f(unsigned short h) {
  return __uint_as_float(((unsigned int)h) << 16);
}

// A fragment (16x32 bf16, row-per-lane): arow points at lane's row + ks + 8*half.
// u[j]: j<4 -> K pair at 2j, j>=4 -> K pair at 2j+8.
__device__ __forceinline__ void load_afrag(ABFrag& f, const unsigned short* arow) {
#pragma unroll
  for (int j = 0; j < 8; ++j) {
    int off = (j < 4) ? (2 * j) : (2 * j + 8);
    f.u[j] = *(const unsigned int*)&arow[off];
  }
}
// B fragment (32x16 bf16, col-per-lane): brow points at lane's column row + ks + 8*half.
// u[j]: j<4 -> K pair at 2j, j>=4 -> K pair at 2j+8.
__device__ __forceinline__ void load_bfrag(ABFrag& f, const unsigned short* brow) {
#pragma unroll
  for (int j = 0; j < 8; ++j) f.u[j] = *(const unsigned int*)&brow[(j < 4) ? (2 * j) : (2 * j + 8)];
}

__global__ void zero_ws_kernel(float* p, int count) {
  int i = blockIdx.x * blockDim.x + threadIdx.x;
  if (i < count) p[i] = 0.0f;
}

// ------------------------------------------------------------------
// Pass 1:  KV[n] += k_hat^T @ v   (128x64 fp32),  ksum[n] += sum_l k_hat[l]
// grid = (L/SEQ_WG, NN), block = 256 (8 wave32)
// ------------------------------------------------------------------
__global__ __launch_bounds__(256) void pass1_kv(const float* __restrict__ kg,
                                                const float* __restrict__ vg,
                                                const float* __restrict__ mask,
                                                float* __restrict__ ws_kv,
                                                float* __restrict__ ws_ksum) {
  const int chunk = blockIdx.x;
  const int n = blockIdx.y;
  const int b = n / HH, h = n % HH;
  const int tid = threadIdx.x;
  const int wave = tid >> 5;
  const int lane = tid & 31;
  const int half = lane >> 4;
  const int l16 = lane & 15;

  __shared__ unsigned short k_lds[FF * KSTR];  // [feature d][seq]  (transposed)
  __shared__ unsigned short v_lds[DD * KSTR];  // [vdim m][seq]     (transposed)

  v8f acc[4] = {};          // wave owns feature tile d0=16*wave, all 4 N tiles
  float ksum_local = 0.0f;  // threads 0..127 own one feature each

  const int l_base = chunk * SEQ_WG;
  const int d0 = wave * 16;

  for (int stage = 0; stage < SEQ_WG / CH; ++stage) {
    const int l0 = l_base + stage * CH;
    __syncthreads();
    // stage CH seq positions: relu, mask, sin/cos featurize, transpose to LDS.
    // float4 global loads -> global_load_b128, fully coalesced.
#pragma unroll
    for (int i = 0; i < (CH * DD) / (256 * 4); ++i) {
      int idx = i * 256 + tid;       // 0..1023
      int ll = idx >> 4;             // 0..63
      int d4 = (idx & 15) << 2;      // 0,4,...,60
      int l = l0 + ll;
      long gidx = ((long)b * LL + l) * EE + h * DD + d4;
      float m = mask[b * LL + l];
      float ang = PI_HALF * (float)(l + 1) / (float)LL;
      float s = __sinf(ang), c = __cosf(ang);
      float4 k4 = *(const float4*)&kg[gidx];
      float4 v4 = *(const float4*)&vg[gidx];
      float ke[4] = {k4.x, k4.y, k4.z, k4.w};
      float ve[4] = {v4.x, v4.y, v4.z, v4.w};
#pragma unroll
      for (int j = 0; j < 4; ++j) {
        float kr = fmaxf(ke[j], 0.0f) * m;
        k_lds[(d4 + j) * KSTR + ll] = f2bf(kr * s);
        k_lds[(d4 + j + DD) * KSTR + ll] = f2bf(kr * c);
        v_lds[(d4 + j) * KSTR + ll] = f2bf(ve[j] * m);
      }
      if (stage + 1 < SEQ_WG / CH) {  // prefetch next stage's lines
        __builtin_prefetch(&kg[gidx + (long)CH * EE], 0, 3);
        __builtin_prefetch(&vg[gidx + (long)CH * EE], 0, 3);
      }
    }
    __syncthreads();
    // ksum partial (feature = tid)
    if (tid < FF) {
#pragma unroll 8
      for (int ll = 0; ll < CH; ++ll) ksum_local += bf2f(k_lds[tid * KSTR + ll]);
    }
    // WMMA accumulate: A = k_hat^T tile (features x seq), B = v tile (seq x vdim).
    // Preload A + all 4 B fragments, then issue the 4 WMMAs back-to-back.
#pragma unroll
    for (int ks = 0; ks < CH; ks += 32) {
      ABFrag fa, fb[4];
      load_afrag(fa, &k_lds[(d0 + l16) * KSTR + ks + half * 8]);
#pragma unroll
      for (int nt = 0; nt < 4; ++nt)
        load_bfrag(fb[nt], &v_lds[(nt * 16 + l16) * KSTR + ks + half * 8]);
#pragma unroll
      for (int nt = 0; nt < 4; ++nt)
        acc[nt] = __builtin_amdgcn_wmma_f32_16x16x32_bf16(
            false, fa.v, false, fb[nt].v, (short)0, acc[nt], false, false);
    }
  }

  // commit partial KV (fp32 atomics into workspace)
#pragma unroll
  for (int nt = 0; nt < 4; ++nt) {
#pragma unroll
    for (int r = 0; r < 8; ++r) {
      int d = d0 + r + half * 8;
      int m = nt * 16 + l16;
      atomicAdd(&ws_kv[((long)n * FF + d) * DD + m], acc[nt][r]);
    }
  }
  if (tid < FF) atomicAdd(&ws_ksum[n * FF + tid], ksum_local);
}

// ------------------------------------------------------------------
// Pass 2:  out[l] = (q_hat[l] @ KV) * z[l],  z = 1/max(q_hat . ksum, eps)
// grid = (L/ROWS, NN), block = 256 (8 wave32)
// ------------------------------------------------------------------
__global__ __launch_bounds__(256) void pass2_out(const float* __restrict__ qg,
                                                 const float* __restrict__ ws_kv,
                                                 const float* __restrict__ ws_ksum,
                                                 float* __restrict__ out) {
  const int tile = blockIdx.x;
  const int n = blockIdx.y;
  const int b = n / HH, h = n % HH;
  const int tid = threadIdx.x;
  const int wave = tid >> 5;
  const int lane = tid & 31;
  const int half = lane >> 4;
  const int l16 = lane & 15;
  const int l0 = tile * ROWS;

  __shared__ unsigned short q_lds[ROWS * QSTR];  // [row][feature] bf16
  __shared__ unsigned short kv_lds[DD * QSTR];   // [vdim m][feature d] bf16 (transposed)
  __shared__ float ksum_lds[FF];
  __shared__ float z_lds[ROWS];

  if (tid < FF) ksum_lds[tid] = ws_ksum[n * FF + tid];

  // KV (fp32, [d][m]) -> bf16 LDS transposed [m][d]; float4 reads (L2-resident)
  for (int i = tid; i < (FF * DD) / 4; i += 256) {
    int d = i >> 4, m4 = (i & 15) << 2;
    float4 c4 = *(const float4*)&ws_kv[((long)n * FF + d) * DD + m4];
    float ce[4] = {c4.x, c4.y, c4.z, c4.w};
#pragma unroll
    for (int j = 0; j < 4; ++j) kv_lds[(m4 + j) * QSTR + d] = f2bf(ce[j]);
  }
  // q tile: relu + sin/cos featurize into LDS; float4 global loads
  for (int i = tid; i < (ROWS * DD) / 4; i += 256) {
    int r = i >> 4, d4 = (i & 15) << 2;
    int l = l0 + r;
    float ang = PI_HALF * (float)(l + 1) / (float)LL;
    float s = __sinf(ang), c = __cosf(ang);
    float4 q4 = *(const float4*)&qg[((long)b * LL + l) * EE + h * DD + d4];
    float qe[4] = {q4.x, q4.y, q4.z, q4.w};
#pragma unroll
    for (int j = 0; j < 4; ++j) {
      float qr = fmaxf(qe[j], 0.0f);
      q_lds[r * QSTR + d4 + j] = f2bf(qr * s);
      q_lds[r * QSTR + DD + d4 + j] = f2bf(qr * c);
    }
  }
  __syncthreads();

  // normalizer z per row
  if (tid < ROWS) {
    float dot = 0.0f;
#pragma unroll 8
    for (int d = 0; d < FF; ++d) dot += bf2f(q_lds[tid * QSTR + d]) * ksum_lds[d];
    z_lds[tid] = 1.0f / fmaxf(dot, EPS);
  }
  __syncthreads();

  // WMMA:  out_tile(16x64) = q_hat(16x128) @ KV(128x64)
  v8f acc[4] = {};
  const int m0 = wave * 16;
#pragma unroll
  for (int ks = 0; ks < FF; ks += 32) {
    ABFrag fa, fb[4];
    load_afrag(fa, &q_lds[(m0 + l16) * QSTR + ks + half * 8]);
#pragma unroll
    for (int nt = 0; nt < 4; ++nt)
      load_bfrag(fb[nt], &kv_lds[(nt * 16 + l16) * QSTR + ks + half * 8]);
#pragma unroll
    for (int nt = 0; nt < 4; ++nt)
      acc[nt] = __builtin_amdgcn_wmma_f32_16x16x32_bf16(
          false, fa.v, false, fb[nt].v, (short)0, acc[nt], false, false);
  }

  // scale by z and store (out shape (B,H,L,D) == (n,L,D))
#pragma unroll
  for (int nt = 0; nt < 4; ++nt) {
#pragma unroll
    for (int r = 0; r < 8; ++r) {
      int row = m0 + r + half * 8;
      int l = l0 + row;
      int m = nt * 16 + l16;
      out[((long)n * LL + l) * DD + m] = acc[nt][r] * z_lds[row];
    }
  }
}

extern "C" void kernel_launch(void* const* d_in, const int* in_sizes, int n_in,
                              void* d_out, int out_size, void* d_ws, size_t ws_size,
                              hipStream_t stream) {
  (void)in_sizes; (void)n_in; (void)out_size; (void)ws_size;
  const float* q = (const float*)d_in[0];
  const float* k = (const float*)d_in[1];
  const float* v = (const float*)d_in[2];
  const float* mask = (const float*)d_in[3];
  float* out = (float*)d_out;

  float* ws_kv = (float*)d_ws;                    // NN*FF*DD fp32 = 2 MB
  float* ws_ksum = ws_kv + (long)NN * FF * DD;    // NN*FF fp32
  int zcount = NN * FF * DD + NN * FF;

  zero_ws_kernel<<<(zcount + 255) / 256, 256, 0, stream>>>(ws_kv, zcount);
  pass1_kv<<<dim3(LL / SEQ_WG, NN), 256, 0, stream>>>(k, v, mask, ws_kv, ws_ksum);
  pass2_out<<<dim3(LL / ROWS, NN), 256, 0, stream>>>(q, ws_kv, ws_ksum, out);
}
